// RoomDecoder_6657199308968
// MI455X (gfx1250) — matrix operand order repaired and checked by reading, not compiled, not run on hardware
//
#include <hip/hip_runtime.h>
#include <hip/hip_bf16.h>
#include <math.h>

// ---------------- compile-time problem constants (match reference) ----------
#define EDIM 512       // embedding/hidden dim E
#define FDIM 7         // room features F
#define NT3  96        // 3E/16 output tiles for gate GEMMs
#define KC_E 16        // E/32  k-chunks
#define KC_2E 32       // 2E/32 k-chunks

typedef __attribute__((ext_vector_type(16))) __bf16 v16bf;
typedef __attribute__((ext_vector_type(8)))  __bf16 v8bf;
typedef __attribute__((ext_vector_type(8)))  float  v8f;
typedef __attribute__((ext_vector_type(4)))  unsigned int u32x4;
typedef __attribute__((ext_vector_type(4)))  int i32x4;
typedef __attribute__((ext_vector_type(8)))  int i32x8;

__device__ __forceinline__ int lane_id() { return (int)(threadIdx.x & 31u); }

__device__ __forceinline__ v8f vzero8() {
    v8f z = {0.f,0.f,0.f,0.f,0.f,0.f,0.f,0.f};
    return z;
}

#if defined(__has_builtin)
#if __has_builtin(__builtin_amdgcn_tensor_load_to_lds) && __has_builtin(__builtin_amdgcn_s_wait_tensorcnt)
#define HAVE_TDM 1
#endif
#endif
#ifndef HAVE_TDM
#define HAVE_TDM 0
#endif

// --- TDM: stage a 16 x EDIM bf16 tile (row stride EDIM) from global to LDS --
// D# per CDNA5 ISA ch.8: group0 = {count/lds_addr/global_addr/type},
// group1 = {data_size=2B, tensor 512 x 16, tile 512 x 16, dim0 stride 512}.
__device__ __forceinline__ void tdm_stage_tile(const __bf16* gsrc, unsigned lds_off) {
#if HAVE_TDM
    unsigned long long ga = (unsigned long long)(size_t)gsrc;
    u32x4 g0;
    g0[0] = 1u;                                     // count=1, user descriptor
    g0[1] = lds_off;                                // lds_addr (bytes)
    g0[2] = (unsigned)(ga & 0xffffffffu);           // global_addr[31:0]
    g0[3] = (unsigned)((ga >> 32) & 0x01ffffffu)    // global_addr[56:32]
          | 0x80000000u;                            // type=2 ("image")
    i32x8 g1;
    g1[0] = (1 << 16);                              // data_size=1 -> 2 bytes
    g1[1] = (EDIM << 16);                           // tensor_dim0 lo16 (=512)
    g1[2] = (16 << 16);                             // dim0 hi=0 | tensor_dim1 lo16 (=16)
    g1[3] = (EDIM << 16);                           // dim1 hi=0 | tile_dim0 (=512)
    g1[4] = 16;                                     // tile_dim1=16, tile_dim2=0
    g1[5] = EDIM;                                   // tensor_dim0_stride lo32 (=512)
    g1[6] = 0;                                      // stride hi | dim1_stride lo
    g1[7] = 0;
    i32x4 z4 = {0,0,0,0};
#if __clang_major__ >= 23
    i32x8 z8 = {0,0,0,0,0,0,0,0};
    __builtin_amdgcn_tensor_load_to_lds(g0, g1, z4, z4, z8, 0);
#else
    __builtin_amdgcn_tensor_load_to_lds(g0, g1, z4, z4, 0);
#endif
#endif
}

__device__ __forceinline__ void tdm_wait() {
#if HAVE_TDM
    __builtin_amdgcn_s_wait_tensorcnt(0);
#endif
}

// Fallback / portable staging: cooperative vectorized copy (all 256 threads)
__device__ __forceinline__ void copy_stage_tile(__bf16* sdst, const __bf16* gsrc) {
    const v8bf* s = (const v8bf*)gsrc;
    v8bf* d = (v8bf*)sdst;
    for (int i = threadIdx.x; i < 16 * EDIM / 8; i += 256) d[i] = s[i];
}

// A fragment (16x32 bf16) from LDS-staged tile, contiguous k order (same as pack_b_kernel):
// lane l: row M=l&15, K range [k + (l>>4)*16 .. +16) contiguous.
__device__ __forceinline__ v16bf load_a_frag_lds(const __bf16* sA, int k) {
    int l = lane_id();
    const __bf16* p = sA + (l & 15) * EDIM + k + ((l >> 4) << 4);
    v8bf lo = *(const v8bf*)(p);
    v8bf hi = *(const v8bf*)(p + 8);
    v16bf a;
#pragma unroll
    for (int i = 0; i < 8; ++i) { a[i] = lo[i]; a[i + 8] = hi[i]; }
    return a;
}

// B fragment (32x16 bf16): packed so each lane's 16 values are contiguous
__device__ __forceinline__ v16bf load_b_frag(const __bf16* __restrict__ pw, int frag) {
    int l = lane_id();
    return *(const v16bf*)(pw + (size_t)frag * 512 + (size_t)l * 16);
}

__device__ __forceinline__ float sigmoidf_(float x) { return 1.f / (1.f + expf(-x)); }

// ---------------- weight packing: fp32 (N x K) row-major -> bf16 B-fragments
// W_cat[n][k] = (k < ksplit) ? W1[n][k] : W2[n][k-ksplit]
__global__ void pack_b_kernel(const float* __restrict__ W1,
                              const float* __restrict__ W2,
                              int ksplit, int K, __bf16* __restrict__ out) {
    int frag = blockIdx.x;
    int KC = K >> 5;
    int nt = frag / KC, kc = frag % KC;
    int l = threadIdx.x;
    int n  = nt * 16 + (l & 15);
    int kb = kc * 32 + ((l >> 4) << 4);
    __bf16* o = out + (size_t)frag * 512 + (size_t)l * 16;
#pragma unroll
    for (int i = 0; i < 16; ++i) {
        int k = kb + i;
        float v = (k < ksplit) ? W1[(size_t)n * ksplit + k]
                               : W2[(size_t)n * (K - ksplit) + (k - ksplit)];
        o[i] = (__bf16)v;
    }
}

__global__ void cvt_bf16_kernel(const float* __restrict__ in,
                                __bf16* __restrict__ out, int n) {
    int i = blockIdx.x * 256 + threadIdx.x;
    if (i < n) out[i] = (__bf16)in[i];
}

// ---------------- GRU layer 0 ----------------------------------------------
__global__ __launch_bounds__(256)
void gru0_kernel(const float* __restrict__ rooms, int step, int SF,
                 const float* __restrict__ w_ih, const float* __restrict__ b_ih,
                 const float* __restrict__ b_hh,
                 const __bf16* __restrict__ pw_hh,
                 float* __restrict__ h0f, __bf16* __restrict__ h0b) {
    __shared__ __bf16 sA[16 * EDIM];     // staged h0 tile (16 KB)
    __shared__ float sx[16][FDIM];
    int mbase = blockIdx.x * 16;
    int tid = threadIdx.x;

#if HAVE_TDM
    if (tid < 32) {
        tdm_stage_tile(h0b + (size_t)mbase * EDIM, (unsigned)(size_t)(void*)sA);
        tdm_wait();
    }
#else
    copy_stage_tile(sA, h0b + (size_t)mbase * EDIM);
#endif
    if (tid < 16 * FDIM) {
        int ml = tid / FDIM, f = tid % FDIM;
        sx[ml][f] = (step == 0) ? 0.f
                  : rooms[(size_t)(mbase + ml) * SF + (size_t)(step - 1) * FDIM + f];
    }
    __syncthreads();

    int w = tid >> 5;
    v8f acc[4][3];
#pragma unroll
    for (int j = 0; j < 4; ++j)
#pragma unroll
        for (int g = 0; g < 3; ++g) acc[j][g] = vzero8();

    for (int kc = 0; kc < KC_E; ++kc) {
        v16bf a = load_a_frag_lds(sA, kc * 32);
        if (kc + 1 < KC_E)
            __builtin_prefetch(pw_hh + (size_t)(w * KC_E + kc + 1) * 512 + lane_id() * 16, 0, 3);
#pragma unroll
        for (int j = 0; j < 4; ++j) {
            int et = w + 8 * j;
#pragma unroll
            for (int g = 0; g < 3; ++g) {
                v16bf b = load_b_frag(pw_hh, (g * 32 + et) * KC_E + kc);
                acc[j][g] = __builtin_amdgcn_wmma_f32_16x16x32_bf16(
                    false, a, false, b, (short)0, acc[j][g], false, false);
            }
        }
    }

    int l = lane_id();
    int nl = l & 15, mo = (l >> 4) << 3;
#pragma unroll
    for (int j = 0; j < 4; ++j) {
        int e = (w + 8 * j) * 16 + nl;
#pragma unroll
        for (int r = 0; r < 8; ++r) {
            int ml = r + mo;
            float gi[3];
#pragma unroll
            for (int g = 0; g < 3; ++g) {
                float s = b_ih[g * EDIM + e];
                const float* wr = w_ih + (size_t)(g * EDIM + e) * FDIM;
#pragma unroll
                for (int f = 0; f < FDIM; ++f) s += sx[ml][f] * wr[f];
                gi[g] = s;
            }
            float ghr = acc[j][0][r] + b_hh[0 * EDIM + e];
            float ghz = acc[j][1][r] + b_hh[1 * EDIM + e];
            float ghn = acc[j][2][r] + b_hh[2 * EDIM + e];
            float rr = sigmoidf_(gi[0] + ghr);
            float zz = sigmoidf_(gi[1] + ghz);
            float nn = tanhf(gi[2] + rr * ghn);
            size_t idx = (size_t)(mbase + ml) * EDIM + e;
            float hold = h0f[idx];
            float hnew = (1.f - zz) * nn + zz * hold;
            h0f[idx] = hnew;
            h0b[idx] = (__bf16)hnew;
        }
    }
}

// ---------------- GRU layer 1 ----------------------------------------------
// s  = [h0_new|h1_old] @ [w_ih_l1; w_hh_l1]^T  (K=1024, all gates)
// gn =  h1_old @ w_hh_l1(n-rows)^T             (K=512, n gate only)
__global__ __launch_bounds__(256)
void gru1_kernel(const float* __restrict__ b_ih, const float* __restrict__ b_hh,
                 const __bf16* __restrict__ pw_cat, const __bf16* __restrict__ pw_hh1,
                 const __bf16* __restrict__ h0b,
                 float* __restrict__ h1f, __bf16* __restrict__ h1b) {
    __shared__ __bf16 sA0[16 * EDIM];    // staged h0_new tile
    __shared__ __bf16 sA1[16 * EDIM];    // staged h1_old tile
    int mbase = blockIdx.x * 16;
    int tid = threadIdx.x;

#if HAVE_TDM
    if (tid < 32) {
        tdm_stage_tile(h0b + (size_t)mbase * EDIM, (unsigned)(size_t)(void*)sA0);
        tdm_stage_tile(h1b + (size_t)mbase * EDIM, (unsigned)(size_t)(void*)sA1);
        tdm_wait();
    }
#else
    copy_stage_tile(sA0, h0b + (size_t)mbase * EDIM);
    copy_stage_tile(sA1, h1b + (size_t)mbase * EDIM);
#endif
    __syncthreads();

    int w = tid >> 5;
    v8f s[4][3], gn[4];
#pragma unroll
    for (int j = 0; j < 4; ++j) {
#pragma unroll
        for (int g = 0; g < 3; ++g) s[j][g] = vzero8();
        gn[j] = vzero8();
    }

    for (int kc = 0; kc < KC_2E; ++kc) {
        v16bf a = load_a_frag_lds((kc < KC_E) ? sA0 : sA1,
                                  ((kc < KC_E) ? kc : kc - KC_E) * 32);
        if (kc + 1 < KC_2E)
            __builtin_prefetch(pw_cat + (size_t)(w * KC_2E + kc + 1) * 512 + lane_id() * 16, 0, 3);
#pragma unroll
        for (int j = 0; j < 4; ++j) {
            int et = w + 8 * j;
#pragma unroll
            for (int g = 0; g < 3; ++g) {
                v16bf b = load_b_frag(pw_cat, (g * 32 + et) * KC_2E + kc);
                s[j][g] = __builtin_amdgcn_wmma_f32_16x16x32_bf16(
                    false, a, false, b, (short)0, s[j][g], false, false);
            }
        }
    }
    for (int kc = 0; kc < KC_E; ++kc) {
        v16bf a = load_a_frag_lds(sA1, kc * 32);
#pragma unroll
        for (int j = 0; j < 4; ++j) {
            int et = w + 8 * j;
            v16bf b = load_b_frag(pw_hh1, (64 + et) * KC_E + kc);
            gn[j] = __builtin_amdgcn_wmma_f32_16x16x32_bf16(
                false, a, false, b, (short)0, gn[j], false, false);
        }
    }

    int l = lane_id();
    int nl = l & 15, mo = (l >> 4) << 3;
#pragma unroll
    for (int j = 0; j < 4; ++j) {
        int e = (w + 8 * j) * 16 + nl;
        float bihn = b_ih[2 * EDIM + e], bhhn = b_hh[2 * EDIM + e];
        float br = b_ih[e] + b_hh[e];
        float bz = b_ih[EDIM + e] + b_hh[EDIM + e];
#pragma unroll
        for (int r = 0; r < 8; ++r) {
            int ml = r + mo;
            float rr = sigmoidf_(s[j][0][r] + br);
            float zz = sigmoidf_(s[j][1][r] + bz);
            float g  = gn[j][r];
            float i_n  = s[j][2][r] - g + bihn;
            float gh_n = g + bhhn;
            float nn = tanhf(i_n + rr * gh_n);
            size_t idx = (size_t)(mbase + ml) * EDIM + e;
            float hold = h1f[idx];
            float hnew = (1.f - zz) * nn + zz * hold;
            h1f[idx] = hnew;
            h1b[idx] = (__bf16)hnew;
        }
    }
}

// ---------------- combiner + LayerNorm + GELU + heads ----------------------
__global__ __launch_bounds__(256)
void comb_kernel(const __bf16* __restrict__ ctxb, const __bf16* __restrict__ h1b,
                 const __bf16* __restrict__ pw_cmb,
                 const float* __restrict__ b_comb,
                 const float* __restrict__ ln_g, const float* __restrict__ ln_b,
                 const float* __restrict__ w_coord, const float* __restrict__ b_coord,
                 const float* __restrict__ w_type,  const float* __restrict__ b_type,
                 const float* __restrict__ w_zone,  const float* __restrict__ b_zone,
                 const float* __restrict__ w_ext,   const float* __restrict__ b_ext,
                 const float* __restrict__ w_stop,  const float* __restrict__ b_stop,
                 float* __restrict__ out, int S_, int T_, int step, int Btot) {
    // 32 KB buffer: first staged A tiles (2 x 16KB bf16), later hid fp32 (16x512)
    __shared__ char smem[16 * EDIM * 4];
    __shared__ float reds[16][16];
    __shared__ float redq[16][16];
    __shared__ float mu_s[16], rs_s[16];

    __bf16* sCtx = (__bf16*)smem;
    __bf16* sH1  = (__bf16*)(smem + 16 * EDIM * 2);
    int mbase = blockIdx.x * 16;
    int tid = threadIdx.x;

#if HAVE_TDM
    if (tid < 32) {
        tdm_stage_tile(ctxb + (size_t)mbase * EDIM, (unsigned)(size_t)(void*)sCtx);
        tdm_stage_tile(h1b  + (size_t)mbase * EDIM, (unsigned)(size_t)(void*)sH1);
        tdm_wait();
    }
#else
    copy_stage_tile(sCtx, ctxb + (size_t)mbase * EDIM);
    copy_stage_tile(sH1,  h1b  + (size_t)mbase * EDIM);
#endif
    __syncthreads();

    int w = tid >> 5;
    v8f acc[4];
#pragma unroll
    for (int j = 0; j < 4; ++j) acc[j] = vzero8();

    for (int kc = 0; kc < KC_2E; ++kc) {
        v16bf a = load_a_frag_lds((kc < KC_E) ? sCtx : sH1,
                                  ((kc < KC_E) ? kc : kc - KC_E) * 32);
        if (kc + 1 < KC_2E)
            __builtin_prefetch(pw_cmb + (size_t)(w * KC_2E + kc + 1) * 512 + lane_id() * 16, 0, 3);
#pragma unroll
        for (int j = 0; j < 4; ++j) {
            v16bf b = load_b_frag(pw_cmb, (w + 8 * j) * KC_2E + kc);
            acc[j] = __builtin_amdgcn_wmma_f32_16x16x32_bf16(
                false, a, false, b, (short)0, acc[j], false, false);
        }
    }
    __syncthreads();                 // staged A dead; reuse smem as fp32 hid

    float* sh = (float*)smem;
    int l = lane_id();
    int nl = l & 15, mo = (l >> 4) << 3;
#pragma unroll
    for (int j = 0; j < 4; ++j) {
        int n = (w + 8 * j) * 16 + nl;
#pragma unroll
        for (int r = 0; r < 8; ++r)
            sh[(r + mo) * EDIM + n] = acc[j][r] + b_comb[n];
    }
    __syncthreads();

    // LayerNorm (biased var, eps=1e-5), then exact-erf GELU
    int row = tid >> 4, seg = tid & 15;
    float s0 = 0.f, s1 = 0.f;
    for (int c = seg * 32; c < seg * 32 + 32; ++c) {
        float v = sh[row * EDIM + c];
        s0 += v; s1 += v * v;
    }
    reds[row][seg] = s0; redq[row][seg] = s1;
    __syncthreads();
    if (seg == 0) {
        float a0 = 0.f, a1 = 0.f;
        for (int i = 0; i < 16; ++i) { a0 += reds[row][i]; a1 += redq[row][i]; }
        float mu = a0 / (float)EDIM;
        float var = a1 / (float)EDIM - mu * mu;
        mu_s[row] = mu;
        rs_s[row] = rsqrtf(var + 1e-5f);
    }
    __syncthreads();
    {
        float mu = mu_s[row], rs = rs_s[row];
        for (int c = seg * 32; c < seg * 32 + 32; ++c) {
            float v = (sh[row * EDIM + c] - mu) * rs * ln_g[c] + ln_b[c];
            v = 0.5f * v * (1.f + erff(v * 0.70710678118654752f));
            sh[row * EDIM + c] = v;
        }
    }
    __syncthreads();

    // heads: 16 rows x 19 outputs
    if (tid < 16 * 19) {
        int rrow = tid / 19, o = tid % 19;
        const float* hw; float hb; size_t base; int width, idx, act;
        size_t BS = (size_t)Btot * S_;
        if (o < 4)       { hw = w_coord + (size_t)o * EDIM; hb = b_coord[o];
                           base = 0;             width = 4;  idx = o;     act = 1; }
        else if (o < 16) { hw = w_type + (size_t)(o - 4) * EDIM; hb = b_type[o - 4];
                           base = BS * 4;        width = T_; idx = o - 4; act = 0; }
        else if (o == 16){ hw = w_zone; hb = b_zone[0];
                           base = BS * (4 + T_); width = 1;  idx = 0;     act = 1; }
        else if (o == 17){ hw = w_ext;  hb = b_ext[0];
                           base = BS * (5 + T_); width = 1;  idx = 0;     act = 1; }
        else             { hw = w_stop; hb = b_stop[0];
                           base = BS * (6 + T_); width = 1;  idx = 0;     act = 0; }
        float d = hb;
        const float* hr = &sh[rrow * EDIM];
        for (int c = 0; c < EDIM; ++c) d += hr[c] * hw[c];
        if (act) d = sigmoidf_(d);
        size_t m = (size_t)(mbase + rrow);
        out[base + (m * S_ + step) * width + idx] = d;
    }
}

// ---------------- host launch ----------------------------------------------
extern "C" void kernel_launch(void* const* d_in, const int* in_sizes, int n_in,
                              void* d_out, int out_size, void* d_ws, size_t ws_size,
                              hipStream_t stream) {
    const float* context = (const float*)d_in[0];
    const float* rooms   = (const float*)d_in[1];
    // d_in[2] = num_rooms (device scalar; S derived from sizes instead)
    const float* w_ih_l0 = (const float*)d_in[3];
    const float* w_hh_l0 = (const float*)d_in[4];
    const float* b_ih_l0 = (const float*)d_in[5];
    const float* b_hh_l0 = (const float*)d_in[6];
    const float* w_ih_l1 = (const float*)d_in[7];
    const float* w_hh_l1 = (const float*)d_in[8];
    const float* b_ih_l1 = (const float*)d_in[9];
    const float* b_hh_l1 = (const float*)d_in[10];
    const float* w_comb  = (const float*)d_in[11];
    const float* b_comb  = (const float*)d_in[12];
    const float* ln_g    = (const float*)d_in[13];
    const float* ln_b    = (const float*)d_in[14];
    const float* w_coord = (const float*)d_in[15];
    const float* b_coord = (const float*)d_in[16];
    const float* w_type  = (const float*)d_in[17];
    const float* b_type  = (const float*)d_in[18];
    const float* w_zone  = (const float*)d_in[19];
    const float* b_zone  = (const float*)d_in[20];
    const float* w_ext   = (const float*)d_in[21];
    const float* b_ext   = (const float*)d_in[22];
    const float* w_stop  = (const float*)d_in[23];
    const float* b_stop  = (const float*)d_in[24];

    int Bn = in_sizes[0] / EDIM;                 // 4096
    int S  = in_sizes[1] / (Bn * FDIM);          // 32
    int T  = in_sizes[17] / EDIM;                // 12
    float* out = (float*)d_out;

    // workspace layout
    char* ws = (char*)d_ws;
    size_t HB  = (size_t)Bn * EDIM * 4;          // fp32 hidden
    size_t HB2 = (size_t)Bn * EDIM * 2;          // bf16 hidden
    float*  h0f  = (float*)(ws);
    float*  h1f  = (float*)(ws + HB);
    __bf16* h0b  = (__bf16*)(ws + 2 * HB);
    __bf16* h1b  = (__bf16*)(ws + 2 * HB + HB2);
    __bf16* ctxb = (__bf16*)(ws + 2 * HB + 2 * HB2);
    char* p = ws + 2 * HB + 3 * HB2;
    __bf16* pw_hh0 = (__bf16*)p; p += (size_t)NT3 * KC_E  * 512 * 2;
    __bf16* pw_cat = (__bf16*)p; p += (size_t)NT3 * KC_2E * 512 * 2;
    __bf16* pw_hh1 = (__bf16*)p; p += (size_t)NT3 * KC_E  * 512 * 2;
    __bf16* pw_cmb = (__bf16*)p; p += (size_t)32  * KC_2E * 512 * 2;

    // zero-init h0/h1 (fp32 + bf16) — deterministic every call
    hipMemsetAsync(ws, 0, 2 * HB + 2 * HB2, stream);

    // per-call packing
    cvt_bf16_kernel<<<(Bn * EDIM + 255) / 256, 256, 0, stream>>>(context, ctxb, Bn * EDIM);
    pack_b_kernel<<<NT3 * KC_E,  32, 0, stream>>>(w_hh_l0, w_hh_l0, EDIM,     EDIM,     pw_hh0);
    pack_b_kernel<<<NT3 * KC_2E, 32, 0, stream>>>(w_ih_l1, w_hh_l1, EDIM,     2 * EDIM, pw_cat);
    pack_b_kernel<<<NT3 * KC_E,  32, 0, stream>>>(w_hh_l1, w_hh_l1, EDIM,     EDIM,     pw_hh1);
    pack_b_kernel<<<32 * KC_2E,  32, 0, stream>>>(w_comb,  w_comb,  2 * EDIM, 2 * EDIM, pw_cmb);

    int nblk = Bn / 16;
    for (int t = 0; t < S; ++t) {
        gru0_kernel<<<nblk, 256, 0, stream>>>(rooms, t, S * FDIM,
                                              w_ih_l0, b_ih_l0, b_hh_l0,
                                              pw_hh0, h0f, h0b);
        gru1_kernel<<<nblk, 256, 0, stream>>>(b_ih_l1, b_hh_l1, pw_cat, pw_hh1,
                                              h0b, h1f, h1b);
        comb_kernel<<<nblk, 256, 0, stream>>>(ctxb, h1b, pw_cmb, b_comb, ln_g, ln_b,
                                              w_coord, b_coord, w_type, b_type,
                                              w_zone, b_zone, w_ext, b_ext,
                                              w_stop, b_stop,
                                              out, S, T, t, Bn);
    }
    (void)n_in; (void)out_size; (void)ws_size;
}
